// CoWindowAttention_80384607912546
// MI455X (gfx1250) — matrix operand order repaired and checked by reading, not compiled, not run on hardware
//
#include <hip/hip_runtime.h>
#include <hip/hip_bf16.h>

// CoWindowAttention fused kernel for gfx1250 (MI455X), round 3.
// Prep kernel: f32->f16 weight conversion + bias_full materialization (workspace).
// Main kernel: one block per window; QKV proj + biased attention (all heads
// batched per phase) + softmax + out proj via v_wmma_f32_16x16x32_f16.

typedef _Float16 half4_t  __attribute__((ext_vector_type(4)));
typedef _Float16 half8_t  __attribute__((ext_vector_type(8)));
typedef _Float16 half16_t __attribute__((ext_vector_type(16)));
typedef float    float8_t __attribute__((ext_vector_type(8)));

#define NWIN   1024
#define NB     256    // big tokens per window
#define NS     64     // small tokens per window
#define CDIM   192
#define NHEAD  6
#define HD     32
#define RT     32     // big-row tile

#define QKVW_ELEMS  (576 * 192)
#define PROJW_ELEMS (192 * 192)
#define BIASF_ELEMS (NHEAD * NB * NS)
#define WS_PROJ_OFF  (QKVW_ELEMS)                 // in f16 elems
#define WS_BIAS_BYTE (2 * (QKVW_ELEMS + PROJW_ELEMS))

#define ZERO8 {0.f,0.f,0.f,0.f,0.f,0.f,0.f,0.f}

__device__ __forceinline__ float8_t wmma16x16x32(half16_t a, half16_t b, float8_t c) {
    return __builtin_amdgcn_wmma_f32_16x16x32_f16(false, a, false, b, (short)0, c, false, false);
}

// A fragment (16x32 f16) from LDS stored row-major (m,k), k contiguous.
// lane m = lane&15, g = lane>>4: K = 16g..16g+15, one contiguous 32B load.
__device__ __forceinline__ half16_t fragA(const _Float16* buf, int stride, int m0, int k0, int lane) {
    int r = lane & 15, g = lane >> 4;
    const _Float16* p = buf + (m0 + r) * stride + k0 + 16 * g;
    return *(const half16_t*)p;
}

// B fragment (32x16 f16) from f16 memory (LDS or global) stored (n,k), k contiguous.
// lane n = lane&15, g = lane>>4: K = 16g..16g+15, one contiguous 32B load.
__device__ __forceinline__ half16_t fragB(const _Float16* buf, int stride, int n0, int k0, int lane) {
    int r = lane & 15, g = lane >> 4;
    return *(const half16_t*)(buf + (n0 + r) * stride + k0 + 16 * g);
}

// ---------------- prep kernel: weights f32->f16, bias_full[h][m][n] ----------------
__global__ __launch_bounds__(256)
void cowin_prep_kernel(const float* __restrict__ qkv_w,
                       const float* __restrict__ proj_w,
                       const float* __restrict__ bias_table,
                       const int*   __restrict__ rel_index,
                       _Float16* __restrict__ w16,         // qkv_w16 then proj_w16
                       float*    __restrict__ bias_full)
{
    int i = blockIdx.x * blockDim.x + threadIdx.x;
    if (i < QKVW_ELEMS) {
        w16[i] = (_Float16)qkv_w[i];
    } else if (i < QKVW_ELEMS + PROJW_ELEMS) {
        w16[i] = (_Float16)proj_w[i - QKVW_ELEMS];
    }
    if (i < BIASF_ELEMS) {
        int h  = i / (NB * NS);
        int mn = i % (NB * NS);
        bias_full[i] = bias_table[rel_index[mn] * NHEAD + h];
    }
}

// ---------------- main fused kernel ----------------
__global__ __launch_bounds__(256)
void cowin_attn_fused_kernel(const float* __restrict__ big_x,
                             const float* __restrict__ small_x,
                             const _Float16* __restrict__ qkv_w16,
                             const float* __restrict__ qkv_b,
                             const _Float16* __restrict__ proj_w16,
                             const float* __restrict__ proj_b,
                             const float* __restrict__ bias_full,
                             float* __restrict__ out)
{
    // 132 KB LDS, explicit layout (all sub-buffers 32B aligned)
    __shared__ __align__(32) char smraw[135168];
    _Float16* Ks = (_Float16*)(smraw);             // [64][192]  K, row=n, k=c contiguous
    _Float16* Vt = (_Float16*)(smraw + 24576);     // [192][64]  V^T, row=c, k=n contiguous
    _Float16* sx = (_Float16*)(smraw + 49152);     // [64][192]  staged small_x (phase 0/1)
    _Float16* xb = (_Float16*)(smraw + 49152);     // [32][192]  staged big_x tile   (alias)
    _Float16* qs = (_Float16*)(smraw + 61440);     // [32][192]  Q tile              (alias)
    _Float16* ps = (_Float16*)(smraw + 49152);     // [6][32][64] probs f16  (aliases xb+qs)
    float*    ss = (float*)   (smraw + 73728);     // [6][32][64] attn logits f32
    _Float16* os = (_Float16*)(smraw + 122880);    // [32][192]  attention output tile

    const int b    = blockIdx.x;
    const int tid  = threadIdx.x;
    const int lane = tid & 31;
    const int w    = __builtin_amdgcn_readfirstlane(tid >> 5);  // scalar wave id 0..7
    const int r    = lane & 15;
    const int g    = lane >> 4;

    // ---- Phase 0: stage small_x[b] (f32 -> f16), vectorized ----
    {
        const float4* s4 = (const float4*)(small_x + (long)b * NS * CDIM);
        for (int i = tid; i < NS * CDIM / 4; i += 256) {
            float4 v = s4[i];
            half4_t h4; h4[0]=(_Float16)v.x; h4[1]=(_Float16)v.y; h4[2]=(_Float16)v.z; h4[3]=(_Float16)v.w;
            *(half4_t*)(sx + i * 4) = h4;
        }
    }
    __syncthreads();

    // ---- Phase 1: K/V = sx @ W_{k,v}^T + b  (V stored transposed) ----
    // 24 j-tiles (cols 192..575) over 8 waves; B-fragments in regs, reused for 4 m-tiles
    for (int jt = w; jt < 24; jt += 8) {
        int j0 = 192 + jt * 16;
        half16_t bf[6];
#pragma unroll
        for (int kt = 0; kt < 6; ++kt) bf[kt] = fragB(qkv_w16, CDIM, j0, kt * 32, lane);
        float bias = qkv_b[j0 + r];
#pragma unroll
        for (int mt = 0; mt < 4; ++mt) {
            float8_t acc = ZERO8;
#pragma unroll
            for (int kt = 0; kt < 6; ++kt)
                acc = wmma16x16x32(fragA(sx, CDIM, mt * 16, kt * 32, lane), bf[kt], acc);
            if (j0 < 384) {         // K part
                int c = j0 - 192 + r;
#pragma unroll
                for (int i = 0; i < 8; ++i)
                    Ks[(mt * 16 + i + 8 * g) * CDIM + c] = (_Float16)(acc[i] + bias);
            } else {                // V part, transposed store
                int c = j0 - 384 + r;
#pragma unroll
                for (int i = 0; i < 8; ++i)
                    Vt[c * NS + (mt * 16 + i + 8 * g)] = (_Float16)(acc[i] + bias);
            }
        }
    }
    __syncthreads();

    // ---- Phase 2: stream big rows in tiles of RT=32 ----
    const float* bxg  = big_x + (long)b * NB * CDIM;
    float*       outg = out   + (long)b * NB * CDIM;

    for (int rb = 0; rb < NB / RT; ++rb) {
        // 2a: stage big_x tile f32 -> f16, vectorized; prefetch next tile into cache
        {
            const float4* b4 = (const float4*)(bxg + rb * RT * CDIM);
            for (int i = tid; i < RT * CDIM / 4; i += 256) {
                float4 v = b4[i];
                half4_t h4; h4[0]=(_Float16)v.x; h4[1]=(_Float16)v.y; h4[2]=(_Float16)v.z; h4[3]=(_Float16)v.w;
                *(half4_t*)(xb + i * 4) = h4;
            }
            if (rb + 1 < NB / RT)
                __builtin_prefetch((const char*)(bxg + (rb + 1) * RT * CDIM) + tid * 96, 0, 3);
        }
        __syncthreads();

        // 2b: Q = xb @ Wq^T + bq (cols 0..191): 12 j-tiles, B-frags reused for 2 m-tiles
        for (int jt = w; jt < 12; jt += 8) {
            int j0 = jt * 16;
            half16_t bf[6];
#pragma unroll
            for (int kt = 0; kt < 6; ++kt) bf[kt] = fragB(qkv_w16, CDIM, j0, kt * 32, lane);
            float bias = qkv_b[j0 + r];
#pragma unroll
            for (int mt = 0; mt < 2; ++mt) {
                float8_t acc = ZERO8;
#pragma unroll
                for (int kt = 0; kt < 6; ++kt)
                    acc = wmma16x16x32(fragA(xb, CDIM, mt * 16, kt * 32, lane), bf[kt], acc);
#pragma unroll
                for (int i = 0; i < 8; ++i)
                    qs[(mt * 16 + i + 8 * g) * CDIM + j0 + r] = (_Float16)(acc[i] + bias);
            }
        }
        __syncthreads();

        // 2c-S: all heads, S = Q_h @ K_h^T + bias_full[h] (6 x 32x64), 48 tiles, K=HD (1 WMMA)
        for (int t = w; t < 48; t += 8) {
            int h = t >> 3, sub = t & 7;
            int mt = sub & 1, nt = sub >> 1;
            float8_t acc = ZERO8;
            acc = wmma16x16x32(fragA(qs, CDIM, mt * 16, h * HD, lane),
                               fragB(Ks, CDIM, nt * 16, h * HD, lane), acc);
            const float* bh = bias_full + (h * NB + rb * RT) * NS;
            float* sh = ss + h * (RT * NS);
#pragma unroll
            for (int i = 0; i < 8; ++i) {
                int ml = mt * 16 + i + 8 * g;
                int n  = nt * 16 + r;
                sh[ml * NS + n] = acc[i] + bh[ml * NS + n];
            }
        }
        __syncthreads();

        // 2c-softmax: 192 rows (6 heads x 32), one row per thread (f32 logits -> f16 probs)
        if (tid < NHEAD * RT) {
            const float* srow = ss + tid * NS;
            _Float16*    prow = ps + tid * NS;
            float mx = -3.4e38f;
#pragma unroll 8
            for (int n = 0; n < NS; ++n) mx = fmaxf(mx, srow[n]);
            float sum = 0.f;
            float e[NS / 8];
            for (int n8 = 0; n8 < NS; n8 += 8) {
#pragma unroll
                for (int n = 0; n < 8; ++n) { e[n % 8] = __expf(srow[n8 + n] - mx); sum += e[n % 8]; }
#pragma unroll
                for (int n = 0; n < 8; ++n) prow[n8 + n] = (_Float16)e[n % 8];
            }
            float inv = 1.f / sum;
            // normalize in f16 (second pass over probs)
#pragma unroll 8
            for (int n = 0; n < NS; ++n) prow[n] = (_Float16)((float)prow[n] * inv);
        }
        __syncthreads();

        // 2c-O: all heads, O_h = P_h @ V_h (6 x 32x32), 24 tiles, K=64 (2 WMMAs)
        for (int t = w; t < 24; t += 8) {
            int h = t >> 2, sub = t & 3;
            int mt = sub & 1, dt = sub >> 1;
            const _Float16* ph = ps + h * (RT * NS);
            float8_t acc = ZERO8;
            acc = wmma16x16x32(fragA(ph, NS, mt * 16, 0, lane),
                               fragB(Vt, NS, h * HD + dt * 16, 0, lane), acc);
            acc = wmma16x16x32(fragA(ph, NS, mt * 16, 32, lane),
                               fragB(Vt, NS, h * HD + dt * 16, 32, lane), acc);
#pragma unroll
            for (int i = 0; i < 8; ++i)
                os[(mt * 16 + i + 8 * g) * CDIM + h * HD + dt * 16 + r] = (_Float16)acc[i];
        }
        __syncthreads();

        // 2d: out = os @ proj_w^T + proj_b, B-frags reused for 2 m-tiles
        for (int jt = w; jt < 12; jt += 8) {
            int j0 = jt * 16;
            half16_t bf[6];
#pragma unroll
            for (int kt = 0; kt < 6; ++kt) bf[kt] = fragB(proj_w16, CDIM, j0, kt * 32, lane);
            float pb = proj_b[j0 + r];
#pragma unroll
            for (int mt = 0; mt < 2; ++mt) {
                float8_t acc = ZERO8;
#pragma unroll
                for (int kt = 0; kt < 6; ++kt)
                    acc = wmma16x16x32(fragA(os, CDIM, mt * 16, kt * 32, lane), bf[kt], acc);
#pragma unroll
                for (int i = 0; i < 8; ++i)
                    outg[(rb * RT + mt * 16 + i + 8 * g) * CDIM + j0 + r] = acc[i] + pb;
            }
        }
        __syncthreads();
    }
}

extern "C" void kernel_launch(void* const* d_in, const int* in_sizes, int n_in,
                              void* d_out, int out_size, void* d_ws, size_t ws_size,
                              hipStream_t stream) {
    (void)in_sizes; (void)n_in; (void)ws_size; (void)out_size;
    const float* big_x      = (const float*)d_in[0];
    const float* small_x    = (const float*)d_in[1];
    const float* qkv_w      = (const float*)d_in[2];
    const float* qkv_b      = (const float*)d_in[3];
    const float* proj_w     = (const float*)d_in[4];
    const float* proj_b     = (const float*)d_in[5];
    const float* bias_table = (const float*)d_in[6];
    const int*   rel_index  = (const int*)d_in[7];
    float* out = (float*)d_out;

    _Float16* w16       = (_Float16*)d_ws;                       // qkv_w16 | proj_w16
    float*    bias_full = (float*)((char*)d_ws + WS_BIAS_BYTE);  // [6][256][64]

    int prep_n = QKVW_ELEMS + PROJW_ELEMS;   // 147456 >= BIASF_ELEMS
    cowin_prep_kernel<<<dim3((prep_n + 255) / 256), dim3(256), 0, stream>>>(
        qkv_w, proj_w, bias_table, rel_index, w16, bias_full);

    cowin_attn_fused_kernel<<<dim3(NWIN), dim3(256), 0, stream>>>(
        big_x, small_x, w16, qkv_b, w16 + WS_PROJ_OFF, proj_b, bias_full, out);
}
